// SelfAttentionBlock_66992899883169
// MI455X (gfx1250) — matrix operand order repaired and checked by reading, not compiled, not run on hardware
//
#include <hip/hip_runtime.h>
#include <hip/hip_bf16.h>

typedef _Float16 half_t;
typedef __attribute__((ext_vector_type(16))) _Float16 v16h;
typedef __attribute__((ext_vector_type(8)))  _Float16 v8h;
typedef __attribute__((ext_vector_type(8)))  float    v8f;
typedef int v4i_ld __attribute__((vector_size(16)));   // matches builtin param

#define BATCH  4
#define CH     256
#define NTOK   4096      // H*W
#define TOK    16384     // BATCH*NTOK
#define DMODEL 256
#define NHEAD  8
#define DHEAD  32

#if defined(__has_builtin)
#if __has_builtin(__builtin_amdgcn_global_load_async_to_lds_b128) && \
    __has_builtin(__builtin_amdgcn_s_wait_asynccnt)
#define HAVE_ASYNC 1
#endif
#endif
#ifndef HAVE_ASYNC
#define HAVE_ASYNC 0
#endif

__device__ inline v8f zero8() {
  v8f z;
#pragma unroll
  for (int i = 0; i < 8; ++i) z[i] = 0.0f;
  return z;
}

// A-matrix fragment (16x32 f16), row-major buffer, row stride ld (halves).
// ISA layout: lanes 0-15 row m hold K 0..7 (V0-3) and 16..23 (V4-7);
// lanes 16-31 hold K 8..15 and 24..31.
__device__ inline v16h load_a_frag(const half_t* base, int ld) {
  const int lane = threadIdx.x & 31;
  const int m    = lane & 15;
  const int koff = (lane < 16) ? 0 : 8;
  const half_t* p = base + m * ld + koff;
  v8h lo = *(const v8h*)(p);
  v8h hi = *(const v8h*)(p + 16);
  return __builtin_shufflevector(lo, hi, 0,1,2,3,4,5,6,7,8,9,10,11,12,13,14,15);
}

// B-matrix fragment (32x16 f16) from an N-major buffer: element (k,n) at
// baseT[n*ld + k]. Lanes 0-15: col n, K 0..7,16..23; lanes 16-31: K 8..15,24..31.
__device__ inline v16h load_b_frag(const half_t* baseT, int ld) {
  const int lane = threadIdx.x & 31;
  const int n    = lane & 15;
  const int koff = (lane < 16) ? 0 : 8;
  const half_t* p = baseT + n * ld + koff;
  v8h lo = *(const v8h*)(p);
  v8h hi = *(const v8h*)(p + 16);
  return __builtin_shufflevector(lo, hi, 0,1,2,3,4,5,6,7,8,9,10,11,12,13,14,15);
}

__device__ inline v8f wmma16(v16h a, v16h b, v8f c) {
  return __builtin_amdgcn_wmma_f32_16x16x32_f16(false, a, false, b, (short)0, c,
                                                false, false);
}

// 16 bytes global -> LDS. Async (CDNA5 GLOBAL_LOAD_ASYNC_TO_LDS_B128) when
// available, otherwise staged through VGPRs.
__device__ inline void stage_b128(const half_t* g, half_t* l) {
#if HAVE_ASYNC
  __builtin_amdgcn_global_load_async_to_lds_b128((v4i_ld*)g, (v4i_ld*)l, 0, 0);
#else
  *(v8h*)l = *(const v8h*)g;
#endif
}
__device__ inline void stage_wait_keep4() {
#if HAVE_ASYNC
  __builtin_amdgcn_s_wait_asynccnt(4);   // leave next tile's 4 copies in flight
#endif
}
__device__ inline void stage_wait_all() {
#if HAVE_ASYNC
  __builtin_amdgcn_s_wait_asynccnt(0);
#endif
}

// ---------------------------------------------------------------------------
// Kernel 1: QKV GEMM.  inp[t][c] = x[b][c][n] (t=b*4096+n); out j in [0,768).
// Writes q16/k16 as [b*8+h][n][dh] f16 and v transposed vT[b*8+h][dh][n] f16.
// ---------------------------------------------------------------------------
__global__ __launch_bounds__(128) void qkv_kernel(
    const float* __restrict__ x, const float* __restrict__ w_qkv,
    const float* __restrict__ b_qkv, half_t* __restrict__ q16,
    half_t* __restrict__ k16, half_t* __restrict__ vT) {
  __shared__ __align__(16) half_t Ash[64 * 40];
  __shared__ __align__(16) half_t Bsh[64 * 40];
  const int tid  = threadIdx.x;
  const int wave = tid >> 5, lane = tid & 31, ln = lane & 15;
  const int t0 = blockIdx.x * 64;   // token tile base
  const int j0 = blockIdx.y * 64;   // output-column tile base
  const int b  = t0 >> 12;
  const int n0 = t0 & (NTOK - 1);

  v8f acc[4];
#pragma unroll
  for (int t = 0; t < 4; ++t) acc[t] = zero8();

  for (int kc = 0; kc < CH; kc += 32) {
    for (int e = tid; e < 64 * 32; e += 128) {     // A tile: x[b][c][n] -> [n][c]
      int cl = e >> 6, nl = e & 63;
      Ash[nl * 40 + cl] = (half_t)x[((b * CH) + kc + cl) * NTOK + n0 + nl];
    }
    for (int e = tid; e < 64 * 32; e += 128) {     // B tile: w_qkv[c][j] -> [j][c]
      int cl = e >> 6, jl = e & 63;
      Bsh[jl * 40 + cl] = (half_t)w_qkv[(kc + cl) * 768 + j0 + jl];
    }
    __syncthreads();
    v16h a = load_a_frag(&Ash[(wave * 16) * 40], 40);
#pragma unroll
    for (int t = 0; t < 4; ++t) {
      v16h bf = load_b_frag(&Bsh[(t * 16) * 40], 40);
      acc[t] = wmma16(a, bf, acc[t]);
    }
    __syncthreads();
  }

#pragma unroll
  for (int t = 0; t < 4; ++t) {
    const int j    = j0 + t * 16 + ln;
    const float bi = b_qkv[j];
    const int sec  = j >> 8;          // 0=q 1=k 2=v
    const int jj   = j & 255;
    const int h = jj >> 5, dh = jj & 31;
#pragma unroll
    for (int r = 0; r < 8; ++r) {
      const int m = wave * 16 + ((lane < 16) ? r : 8 + r);
      const int n = n0 + m;
      half_t hv = (half_t)(acc[t][r] + bi);
      if (sec == 0)      q16[((b * NHEAD + h) * NTOK + n) * DHEAD + dh] = hv;
      else if (sec == 1) k16[((b * NHEAD + h) * NTOK + n) * DHEAD + dh] = hv;
      else               vT[((b * NHEAD + h) * DHEAD + dh) * NTOK + n]  = hv;
    }
  }
}

// ---------------------------------------------------------------------------
// Kernel 2: flash attention with double-buffered async K/V staging.
// Workgroup = 4 waves = 64 query rows; key blocks of 64.
// Per block: 4 WMMA (S = Q K^T), online softmax, P via per-wave LDS slab,
// 4 WMMA (P V).  grid = (N/64, B*NHEAD), 128 threads.
// ---------------------------------------------------------------------------
__global__ __launch_bounds__(128) void attn_kernel(
    const half_t* __restrict__ q16, const half_t* __restrict__ k16,
    const half_t* __restrict__ vT, half_t* __restrict__ o16) {
  __shared__ __align__(16) half_t Ksh[2][64 * 40];   // [key][dh], padded
  __shared__ __align__(16) half_t Vsh[2][32 * 72];   // [dh][key], padded
  __shared__ __align__(16) half_t Psh[4][16 * 72];   // per-wave P slab
  const int tid = threadIdx.x, wave = tid >> 5, lane = tid & 31, ln = lane & 15;
  const int bh  = blockIdx.y;
  const int qr0 = blockIdx.x * 64 + wave * 16;
  const float scale = 0.17677669529663687f;  // 1/sqrt(32)

  const half_t* kbase = k16 + (size_t)bh * NTOK * DHEAD;
  const half_t* vbase = vT + (size_t)bh * DHEAD * NTOK;

  const v16h aq = load_a_frag(q16 + (bh * NTOK + qr0) * DHEAD, DHEAD);
  v8f acc0 = zero8(), acc1 = zero8();
  const v8f z = zero8();
  float mrow[8], lrow[8];
#pragma unroll
  for (int r = 0; r < 8; ++r) { mrow[r] = -1e30f; lrow[r] = 0.0f; }
  half_t* pw = &Psh[wave][0];

  // Cooperative stage of one 64-key K tile + V tile (2+2 b128 per thread).
  auto issue = [&](int buf, int kb) {
#pragma unroll
    for (int i = 0; i < 2; ++i) {                 // K: 256 x 16B chunks
      int e = tid + i * 128;
      int row = e >> 2, c = e & 3;
      stage_b128(kbase + (size_t)(kb + row) * DHEAD + c * 8,
                 &Ksh[buf][row * 40 + c * 8]);
    }
#pragma unroll
    for (int i = 0; i < 2; ++i) {                 // V: 256 x 16B chunks
      int e = tid + i * 128;
      int dh = e >> 3, c = e & 7;
      stage_b128(vbase + (size_t)dh * NTOK + kb + c * 8,
                 &Vsh[buf][dh * 72 + c * 8]);
    }
  };

  issue(0, 0);
  for (int it = 0; it < NTOK / 64; ++it) {
    const int kb  = it * 64;
    const int buf = it & 1;
    if (kb + 64 < NTOK) { issue(buf ^ 1, kb + 64); stage_wait_keep4(); }
    else                { stage_wait_all(); }
    __syncthreads();

    v16h bk0 = load_b_frag(&Ksh[buf][0],       40);
    v16h bk1 = load_b_frag(&Ksh[buf][16 * 40], 40);
    v16h bk2 = load_b_frag(&Ksh[buf][32 * 40], 40);
    v16h bk3 = load_b_frag(&Ksh[buf][48 * 40], 40);
    v8f s0 = wmma16(aq, bk0, z);
    v8f s1 = wmma16(aq, bk1, z);
    v8f s2 = wmma16(aq, bk2, z);
    v8f s3 = wmma16(aq, bk3, z);

#pragma unroll
    for (int r = 0; r < 8; ++r) {
      float a0 = s0[r] * scale, a1 = s1[r] * scale;
      float a2 = s2[r] * scale, a3 = s3[r] * scale;
      float mx = fmaxf(fmaxf(a0, a1), fmaxf(a2, a3));
      mx = fmaxf(mx, __shfl_xor(mx, 1, 32));
      mx = fmaxf(mx, __shfl_xor(mx, 2, 32));
      mx = fmaxf(mx, __shfl_xor(mx, 4, 32));
      mx = fmaxf(mx, __shfl_xor(mx, 8, 32));
      float mn = fmaxf(mrow[r], mx);
      float al = __expf(mrow[r] - mn);
      float e0 = __expf(a0 - mn), e1 = __expf(a1 - mn);
      float e2 = __expf(a2 - mn), e3 = __expf(a3 - mn);
      float rs = (e0 + e1) + (e2 + e3);
      rs += __shfl_xor(rs, 1, 32);
      rs += __shfl_xor(rs, 2, 32);
      rs += __shfl_xor(rs, 4, 32);
      rs += __shfl_xor(rs, 8, 32);
      lrow[r] = lrow[r] * al + rs;
      mrow[r] = mn;
      acc0[r] *= al; acc1[r] *= al;
      const int m = (lane < 16) ? r : 8 + r;     // C-layout -> LDS row-major
      pw[m * 72 + ln]      = (half_t)e0;
      pw[m * 72 + 16 + ln] = (half_t)e1;
      pw[m * 72 + 32 + ln] = (half_t)e2;
      pw[m * 72 + 48 + ln] = (half_t)e3;
    }

    v16h ap0  = load_a_frag(pw, 72);             // P keys 0..31
    v16h ap1  = load_a_frag(pw + 32, 72);        // P keys 32..63
    v16h bv00 = load_b_frag(&Vsh[buf][0], 72);           // d 0..15, k 0..31
    v16h bv01 = load_b_frag(&Vsh[buf][32], 72);          // d 0..15, k 32..63
    v16h bv10 = load_b_frag(&Vsh[buf][16 * 72], 72);     // d 16..31, k 0..31
    v16h bv11 = load_b_frag(&Vsh[buf][16 * 72 + 32], 72);
    acc0 = wmma16(ap0, bv00, acc0);
    acc0 = wmma16(ap1, bv01, acc0);
    acc1 = wmma16(ap0, bv10, acc1);
    acc1 = wmma16(ap1, bv11, acc1);
    __syncthreads();   // all waves done with buf before it is re-staged
  }

  const int b = bh >> 3, h = bh & 7;
#pragma unroll
  for (int r = 0; r < 8; ++r) {
    const float inv = 1.0f / lrow[r];
    const int m = (lane < 16) ? r : 8 + r;
    const int n = qr0 + m;
    o16[((size_t)(b * NTOK + n)) * DMODEL + h * DHEAD + ln]      = (half_t)(acc0[r] * inv);
    o16[((size_t)(b * NTOK + n)) * DMODEL + h * DHEAD + 16 + ln] = (half_t)(acc1[r] * inv);
  }
}

// ---------------------------------------------------------------------------
// Kernel 3: out-proj GEMM + bias + residual; writes y=[b][d][n] f32 and
// accumulates per-(b,group) sum / sumsq via LDS + global atomics.
// ---------------------------------------------------------------------------
__global__ __launch_bounds__(128) void proj_kernel(
    const half_t* __restrict__ o16, const float* __restrict__ w_proj,
    const float* __restrict__ b_proj, const float* __restrict__ x,
    float* __restrict__ y, float* __restrict__ gstats) {
  __shared__ __align__(16) half_t Bsh[64 * 40];
  __shared__ float red[16];
  const int tid = threadIdx.x, wave = tid >> 5, lane = tid & 31, ln = lane & 15;
  const int t0  = blockIdx.x * 64;
  const int dp0 = blockIdx.y * 64;
  const int b = t0 >> 12, n0 = t0 & (NTOK - 1);
  if (tid < 16) red[tid] = 0.0f;

  v8f acc[4];
#pragma unroll
  for (int t = 0; t < 4; ++t) acc[t] = zero8();

  for (int kc = 0; kc < DMODEL; kc += 32) {
    for (int e = tid; e < 64 * 32; e += 128) {   // w_proj[c][dp] -> [dp][c] f16
      int cl = e >> 6, jl = e & 63;
      Bsh[jl * 40 + cl] = (half_t)w_proj[(kc + cl) * DMODEL + dp0 + jl];
    }
    __syncthreads();
    v16h a = load_a_frag(o16 + (size_t)(t0 + wave * 16) * DMODEL + kc, DMODEL);
#pragma unroll
    for (int t = 0; t < 4; ++t) {
      v16h bf = load_b_frag(&Bsh[(t * 16) * 40], 40);
      acc[t] = wmma16(a, bf, acc[t]);
    }
    __syncthreads();
  }

#pragma unroll
  for (int t = 0; t < 4; ++t) {
    const int dp   = dp0 + t * 16 + ln;
    const float bi = b_proj[dp];
    float ls = 0.0f, ls2 = 0.0f;
#pragma unroll
    for (int r = 0; r < 8; ++r) {
      const int m = wave * 16 + ((lane < 16) ? r : 8 + r);
      const int n = n0 + m;
      const size_t idx = (size_t)(b * CH + dp) * NTOK + n;
      float val = acc[t][r] + bi + x[idx];
      y[idx] = val;
      ls += val; ls2 += val * val;
    }
    const int lg = (t * 16 + ln) >> 3;   // local group 0..7
    atomicAdd(&red[lg * 2 + 0], ls);
    atomicAdd(&red[lg * 2 + 1], ls2);
  }
  __syncthreads();
  if (tid < 8) {
    const int g = (dp0 >> 3) + tid;
    atomicAdd(&gstats[(b * 32 + g) * 2 + 0], red[tid * 2 + 0]);
    atomicAdd(&gstats[(b * 32 + g) * 2 + 1], red[tid * 2 + 1]);
  }
}

// ---------------------------------------------------------------------------
// Kernel 4: GroupNorm apply. One block per (b,c) channel row of 4096.
// ---------------------------------------------------------------------------
__global__ __launch_bounds__(256) void gn_kernel(
    const float* __restrict__ y, const float* __restrict__ gstats,
    const float* __restrict__ gamma, const float* __restrict__ beta,
    float* __restrict__ out) {
  const int bc = blockIdx.x;
  const int b = bc >> 8, c = bc & 255;
  const int g = c >> 3;
  const float cnt  = 8.0f * (float)NTOK;
  const float s1   = gstats[(b * 32 + g) * 2 + 0];
  const float s2   = gstats[(b * 32 + g) * 2 + 1];
  const float mean = s1 / cnt;
  const float var  = s2 / cnt - mean * mean;
  const float rstd = rsqrtf(var + 1e-5f);
  const float ga = gamma[c], be = beta[c];
  const float4* yp = (const float4*)(y + (size_t)(b * CH + c) * NTOK);
  float4* op = (float4*)(out + (size_t)(b * CH + c) * NTOK);
  for (int i = threadIdx.x; i < NTOK / 4; i += 256) {
    float4 t = yp[i];
    t.x = (t.x - mean) * rstd * ga + be;
    t.y = (t.y - mean) * rstd * ga + be;
    t.z = (t.z - mean) * rstd * ga + be;
    t.w = (t.w - mean) * rstd * ga + be;
    op[i] = t;
  }
}

extern "C" void kernel_launch(void* const* d_in, const int* in_sizes, int n_in,
                              void* d_out, int out_size, void* d_ws,
                              size_t ws_size, hipStream_t stream) {
  const float* x      = (const float*)d_in[0];
  const float* w_qkv  = (const float*)d_in[1];
  const float* b_qkv  = (const float*)d_in[2];
  const float* w_proj = (const float*)d_in[3];
  const float* b_proj = (const float*)d_in[4];
  const float* gamma  = (const float*)d_in[5];
  const float* beta   = (const float*)d_in[6];
  float* out = (float*)d_out;

  char* ws = (char*)d_ws;
  const size_t MB = 1u << 20;
  half_t* q16 = (half_t*)(ws + 0 * MB);    // 8 MB
  half_t* k16 = (half_t*)(ws + 8 * MB);    // 8 MB
  half_t* vT  = (half_t*)(ws + 16 * MB);   // 8 MB
  half_t* o16 = (half_t*)(ws + 24 * MB);   // 8 MB
  float*  y   = (float*)(ws + 0 * MB);     // 16 MB, reuses q16/k16 (dead by then)
  float*  st  = (float*)(ws + 32 * MB);    // 1 KB stats

  (void)hipMemsetAsync(st, 0, BATCH * 32 * 2 * sizeof(float), stream);
  qkv_kernel<<<dim3(TOK / 64, 768 / 64), 128, 0, stream>>>(x, w_qkv, b_qkv,
                                                           q16, k16, vT);
  attn_kernel<<<dim3(NTOK / 64, BATCH * NHEAD), 128, 0, stream>>>(q16, k16, vT,
                                                                  o16);
  proj_kernel<<<dim3(TOK / 64, DMODEL / 64), 128, 0, stream>>>(o16, w_proj,
                                                               b_proj, x, y, st);
  gn_kernel<<<dim3(BATCH * CH), 256, 0, stream>>>(y, st, gamma, beta, out);
}
